// CausalSelfAttention_80109730005655
// MI455X (gfx1250) — matrix operand order repaired and checked by reading, not compiled, not run on hardware
//
#include <hip/hip_runtime.h>

// ---------------------------------------------------------------------------
// Types for CDNA5 WMMA (wave32, 16x16x32 bf16 -> f32)
// ---------------------------------------------------------------------------
typedef __bf16 bf16_t;
typedef __attribute__((ext_vector_type(8)))  __bf16 v8bf;
typedef __attribute__((ext_vector_type(16))) __bf16 v16bf;
typedef __attribute__((ext_vector_type(8)))  float  v8f;

struct alignas(16) B16x16 { bf16_t v[16]; };

// A-matrix fragment (16x32 bf16): lane M = lane&15, element e -> K = e + 16*(lane>=16)
static __device__ inline v16bf lds_load_a_frag(const bf16_t* p) {
  const v8bf* q = (const v8bf*)p;
  v8bf lo = q[0];   // K: base .. base+7
  v8bf hi = q[1];   // K: base+8 .. base+15
  return __builtin_shufflevector(lo, hi, 0,1,2,3,4,5,6,7,8,9,10,11,12,13,14,15);
}

// B-matrix fragment (32x16 bf16): lane N = lane&15, element e -> K = e + 16*(lane>=16)
static __device__ inline v16bf lds_load_b_frag(const bf16_t* p) {
  const v8bf* q = (const v8bf*)p;
  return __builtin_shufflevector(q[0], q[1], 0,1,2,3,4,5,6,7,8,9,10,11,12,13,14,15);
}

static __device__ inline v8f wmma_bf16(v16bf a, v16bf b, v8f c) {
  return __builtin_amdgcn_wmma_f32_16x16x32_bf16(false, a, false, b, (short)0, c, false, false);
}

// ---------------------------------------------------------------------------
// CDNA5 async global->LDS DMA (ASYNCcnt-tracked), 16B per active lane.
// the LDS address is taken from the vdst VGPR; flat LDS pointers carry the LDS offset in
// their low 32 bits, so a truncating cast yields the required VGPR value.
// ---------------------------------------------------------------------------
static __device__ inline void async_copy_16B(unsigned lds_off, const void* gaddr) {
  asm volatile("global_load_async_to_lds_b128 %0, %1, off"
               :: "v"(lds_off), "v"(gaddr)
               : "memory");
}
static __device__ inline void wait_asynccnt0() {
  asm volatile("s_wait_asynccnt 0" ::: "memory");
}

// ---------------------------------------------------------------------------
// fp32 -> bf16 conversion (memory-trivial)
// ---------------------------------------------------------------------------
__global__ void f32_to_bf16_kernel(const float* __restrict__ in, bf16_t* __restrict__ out, int n) {
  for (int i = blockIdx.x * blockDim.x + threadIdx.x; i < n; i += gridDim.x * blockDim.x)
    out[i] = (bf16_t)in[i];
}

// ---------------------------------------------------------------------------
// GEMM: out[M,N] = A[M,K] (bf16) * B[K,N] (bf16) + bias[N]
// Block: 256 threads = 8 waves, 128x128 tile, wave tile 32x64, K-step 32.
// Double-buffered LDS; A staged by async DMA, B transposed through registers.
// One barrier per K-iteration.
// ---------------------------------------------------------------------------
template<bool OUTBF>
__global__ __launch_bounds__(256) void gemm_bias_kernel(
    const bf16_t* __restrict__ A, const bf16_t* __restrict__ Bm,
    const float* __restrict__ bias, void* __restrict__ outP,
    int M, int N, int K)
{
  __shared__ bf16_t As[2][128][40];   // [buf][m][k], stride 80B (16B aligned rows)
  __shared__ bf16_t Bs[2][128][40];   // [buf][n][k], B transposed

  const int tid    = threadIdx.x;
  const int lane   = tid & 31;
  const int wave   = tid >> 5;
  const int wm     = wave & 3;     // M sub-tile (32 rows each)
  const int wn     = wave >> 2;    // N sub-tile (64 cols each)
  const int laneLo = lane & 15;
  const int laneHi = lane >> 4;

  const int bm = blockIdx.y * 128;
  const int bn = blockIdx.x * 128;

  const int brow = tid >> 3;           // 0..31 (k)
  const int bseg = (tid & 7) * 16;     // 0..112 (n)

  auto asyncA = [&](int kt, int buf) {   // 128x32 tile: 512 x 16B chunks, 2 per thread
    const int k0 = kt << 5;
#pragma unroll
    for (int j = 0; j < 2; ++j) {
      const int c = tid + j * 256;
      const int row = c >> 2, chunk = (c & 3) * 8;
      async_copy_16B((unsigned)(uintptr_t)&As[buf][row][chunk],
                     A + (size_t)(bm + row) * K + k0 + chunk);
    }
  };
  auto loadBreg = [&](int kt) {
    return *(const B16x16*)(Bm + (size_t)((kt << 5) + brow) * N + bn + bseg);
  };
  auto storeB = [&](const B16x16& bt, int buf) {
#pragma unroll
    for (int i = 0; i < 16; ++i) Bs[buf][bseg + i][brow] = bt.v[i];
  };

  // prologue: stage tile 0
  asyncA(0, 0);
  B16x16 bt = loadBreg(0);
  storeB(bt, 0);
  wait_asynccnt0();
  __syncthreads();

  v8f acc[2][4] = {};
  int cur = 0;
  const int ktiles = K >> 5;
  for (int kt = 0; kt < ktiles; ++kt) {
    const bool hasNext = (kt + 1 < ktiles);
    if (hasNext) {               // issue next tile's loads before computing
      asyncA(kt + 1, cur ^ 1);
      bt = loadBreg(kt + 1);
    }

    v16bf afrag[2], bfrag[4];
#pragma unroll
    for (int mi = 0; mi < 2; ++mi)
      afrag[mi] = lds_load_a_frag(&As[cur][wm * 32 + mi * 16 + laneLo][laneHi * 16]);
#pragma unroll
    for (int ni = 0; ni < 4; ++ni)
      bfrag[ni] = lds_load_b_frag(&Bs[cur][wn * 64 + ni * 16 + laneLo][laneHi * 16]);
#pragma unroll
    for (int mi = 0; mi < 2; ++mi)
#pragma unroll
      for (int ni = 0; ni < 4; ++ni)
        acc[mi][ni] = wmma_bf16(afrag[mi], bfrag[ni], acc[mi][ni]);

    if (hasNext) {
      storeB(bt, cur ^ 1);
      wait_asynccnt0();
      __syncthreads();
      cur ^= 1;
    }
  }

  // epilogue
  float bv[4];
#pragma unroll
  for (int ni = 0; ni < 4; ++ni) bv[ni] = bias[bn + wn * 64 + ni * 16 + laneLo];
#pragma unroll
  for (int mi = 0; mi < 2; ++mi)
#pragma unroll
    for (int r = 0; r < 8; ++r) {
      const int row = bm + wm * 32 + mi * 16 + r + laneHi * 8;
#pragma unroll
      for (int ni = 0; ni < 4; ++ni) {
        const int col = bn + wn * 64 + ni * 16 + laneLo;
        const float v = acc[mi][ni][r] + bv[ni];
        if (OUTBF) ((bf16_t*)outP)[(size_t)row * N + col] = (bf16_t)v;
        else       ((float*) outP)[(size_t)row * N + col] = v;
      }
    }
}

// ---------------------------------------------------------------------------
// Flash attention: grid (T/128, B*H), block 256 threads (8 waves).
// 128-query tile per block; each wave owns 16 query rows; streams 64-key tiles
// with causal per-wave early-out. Q/K staged by async DMA, V transposed.
// ---------------------------------------------------------------------------
#define T_SEQ 2048
#define C_DIM 1024
#define H_NUM 16
#define D_HEAD 64
#define QTILE 128

__global__ __launch_bounds__(256) void attn_kernel(
    const bf16_t* __restrict__ qkv, bf16_t* __restrict__ y)
{
  __shared__ bf16_t Qs[QTILE][72];    // [q][d]
  __shared__ bf16_t Ks[64][72];       // [key][d]  (row-major == K^T B-operand)
  __shared__ bf16_t Vt[64][72];       // [d][key]  (V transposed for PV B-operand)
  __shared__ bf16_t Ps[8][16][72];    // per-wave P staging (C-layout -> A-layout)

  const int tid    = threadIdx.x;
  const int lane   = tid & 31;
  const int wave   = tid >> 5;
  const int laneLo = lane & 15;
  const int laneHi = lane >> 4;

  const int qt = blockIdx.x;
  const int b  = blockIdx.y >> 4;
  const int h  = blockIdx.y & 15;
  const int q0 = qt * QTILE;

  const size_t rowStride = 3 * C_DIM;
  const bf16_t* base = qkv + (size_t)(b * T_SEQ) * rowStride;
  const int qcol = h * D_HEAD;
  const int kcol = C_DIM + h * D_HEAD;
  const int vcol = 2 * C_DIM + h * D_HEAD;

  // async-stage Q tile (128x64): 1024 x 16B chunks, 4 per thread
#pragma unroll
  for (int j = 0; j < 4; ++j) {
    const int c = tid + j * 256;
    const int row = c >> 3, chunk = (c & 7) * 8;
    async_copy_16B((unsigned)(uintptr_t)&Qs[row][chunk],
                   base + (size_t)(q0 + row) * rowStride + qcol + chunk);
  }

  v8f o[4] = {};
  float mrun[8], lrun[8];
#pragma unroll
  for (int r = 0; r < 8; ++r) { mrun[r] = -1e30f; lrun[r] = 0.f; }

  const int qw0  = q0 + wave * 16;               // wave's first query row
  const int kmax = (q0 + QTILE - 1) >> 6;        // last key tile index
  for (int kt = 0; kt <= kmax; ++kt) {
    __syncthreads();   // previous iteration's consumers done
    // async-stage K tile (64x64): 512 x 16B chunks, 2 per thread
#pragma unroll
    for (int j = 0; j < 2; ++j) {
      const int c = tid + j * 256;
      const int row = c >> 3, chunk = (c & 7) * 8;
      async_copy_16B((unsigned)(uintptr_t)&Ks[row][chunk],
                     base + (size_t)(kt * 64 + row) * rowStride + kcol + chunk);
    }
    // V tile transposed through registers
    {
      const int row = tid >> 2;
      const int seg = (tid & 3) * 16;
      B16x16 vv = *(const B16x16*)(base + (size_t)(kt * 64 + row) * rowStride + vcol + seg);
#pragma unroll
      for (int i = 0; i < 16; ++i) Vt[seg + i][row] = vv.v[i];
    }
    wait_asynccnt0();   // covers K (and Q on the first iteration)
    __syncthreads();

    if (kt * 64 > qw0 + 15) continue;            // wave fully above diagonal
    const bool needMask = (kt * 64 + 63 > qw0);

    // S = Q K^T for this wave's 16 rows (fp32 accum); hoist B-frags per k-step
    v8f s[4] = {};
#pragma unroll
    for (int ks = 0; ks < 2; ++ks) {
      v16bf aq = lds_load_a_frag(&Qs[wave * 16 + laneLo][ks * 32 + laneHi * 16]);
      v16bf bk[4];
#pragma unroll
      for (int ni = 0; ni < 4; ++ni)
        bk[ni] = lds_load_b_frag(&Ks[ni * 16 + laneLo][ks * 32 + laneHi * 16]);
#pragma unroll
      for (int ni = 0; ni < 4; ++ni)
        s[ni] = wmma_bf16(aq, bk[ni], s[ni]);
    }

    // scale + causal mask
    const float scale = 0.125f;  // 1/sqrt(64)
#pragma unroll
    for (int ni = 0; ni < 4; ++ni)
#pragma unroll
      for (int r = 0; r < 8; ++r) {
        float v = s[ni][r] * scale;
        if (needMask) {
          const int key  = kt * 64 + ni * 16 + laneLo;
          const int qrow = qw0 + r + laneHi * 8;
          if (key > qrow) v = -1e30f;
        }
        s[ni][r] = v;
      }

    // row max across the 16 lanes sharing a row set
    float rmax[8];
#pragma unroll
    for (int r = 0; r < 8; ++r) {
      float v = fmaxf(fmaxf(s[0][r], s[1][r]), fmaxf(s[2][r], s[3][r]));
      v = fmaxf(v, __shfl_xor(v, 1, 32));
      v = fmaxf(v, __shfl_xor(v, 2, 32));
      v = fmaxf(v, __shfl_xor(v, 4, 32));
      v = fmaxf(v, __shfl_xor(v, 8, 32));
      rmax[r] = v;
    }

    float alpha[8];
#pragma unroll
    for (int r = 0; r < 8; ++r) {
      const float mn = fmaxf(mrun[r], rmax[r]);
      alpha[r] = __expf(mrun[r] - mn);
      mrun[r]  = mn;
    }

    // P = exp(S - m) -> per-wave LDS (C-layout -> A-layout), rescale O
    float rsum[8] = {};
#pragma unroll
    for (int ni = 0; ni < 4; ++ni)
#pragma unroll
      for (int r = 0; r < 8; ++r) {
        const float p = __expf(s[ni][r] - mrun[r]);
        rsum[r] += p;
        Ps[wave][r + laneHi * 8][ni * 16 + laneLo] = (bf16_t)p;
        o[ni][r] *= alpha[r];
      }
#pragma unroll
    for (int r = 0; r < 8; ++r) {
      float v = rsum[r];
      v += __shfl_xor(v, 1, 32);
      v += __shfl_xor(v, 2, 32);
      v += __shfl_xor(v, 4, 32);
      v += __shfl_xor(v, 8, 32);
      lrun[r] = lrun[r] * alpha[r] + v;
    }

    // O += P V  (reload P as A-fragments; hoist V B-frags)
#pragma unroll
    for (int ks = 0; ks < 2; ++ks) {
      v16bf ap = lds_load_a_frag(&Ps[wave][laneLo][ks * 32 + laneHi * 16]);
      v16bf bv[4];
#pragma unroll
      for (int ni = 0; ni < 4; ++ni)
        bv[ni] = lds_load_b_frag(&Vt[ni * 16 + laneLo][ks * 32 + laneHi * 16]);
#pragma unroll
      for (int ni = 0; ni < 4; ++ni)
        o[ni] = wmma_bf16(ap, bv[ni], o[ni]);
    }
  }

  // normalize and write y (bf16, [B,T,C] with col = h*64 + d)
#pragma unroll
  for (int r = 0; r < 8; ++r) {
    const float inv = 1.0f / lrun[r];
    const int qrow = q0 + wave * 16 + r + laneHi * 8;
    const size_t rowOff = (size_t)(b * T_SEQ + qrow) * C_DIM + h * D_HEAD;
#pragma unroll
    for (int ni = 0; ni < 4; ++ni)
      y[rowOff + ni * 16 + laneLo] = (bf16_t)(o[ni][r] * inv);
  }
}

// ---------------------------------------------------------------------------
// Launch
// ---------------------------------------------------------------------------
extern "C" void kernel_launch(void* const* d_in, const int* in_sizes, int n_in,
                              void* d_out, int out_size, void* d_ws, size_t ws_size,
                              hipStream_t stream) {
  (void)in_sizes; (void)n_in; (void)out_size; (void)ws_size;

  const float* x      = (const float*)d_in[0];
  const float* w_attn = (const float*)d_in[1];
  const float* b_attn = (const float*)d_in[2];
  const float* w_proj = (const float*)d_in[3];
  const float* b_proj = (const float*)d_in[4];

  const int Bb = 4, T = T_SEQ, C = C_DIM;
  const int M = Bb * T;                       // 8192
  const size_t nX   = (size_t)M * C;
  const size_t nWA  = (size_t)C * 3 * C;
  const size_t nWP  = (size_t)C * C;
  const size_t nQKV = (size_t)M * 3 * C;
  const size_t nY   = (size_t)M * C;

  char* ws = (char*)d_ws;
  auto alloc = [&](size_t bytes) { char* p = ws; ws += (bytes + 255) & ~(size_t)255; return p; };
  bf16_t* xb  = (bf16_t*)alloc(nX * 2);
  bf16_t* wab = (bf16_t*)alloc(nWA * 2);
  bf16_t* wpb = (bf16_t*)alloc(nWP * 2);
  bf16_t* qkv = (bf16_t*)alloc(nQKV * 2);
  bf16_t* yb  = (bf16_t*)alloc(nY * 2);

  f32_to_bf16_kernel<<<2048, 256, 0, stream>>>(x, xb, (int)nX);
  f32_to_bf16_kernel<<<2048, 256, 0, stream>>>(w_attn, wab, (int)nWA);
  f32_to_bf16_kernel<<<1024, 256, 0, stream>>>(w_proj, wpb, (int)nWP);

  // qkv = x @ w_attn + b_attn  -> bf16
  gemm_bias_kernel<true><<<dim3(3 * C / 128, M / 128), 256, 0, stream>>>(
      xb, wab, b_attn, (void*)qkv, M, 3 * C, C);

  // flash attention -> y (bf16)
  attn_kernel<<<dim3(T / QTILE, Bb * H_NUM), 256, 0, stream>>>(qkv, yb);

  // out = y @ w_proj + b_proj  -> fp32 d_out
  gemm_bias_kernel<false><<<dim3(C / 128, M / 128), 256, 0, stream>>>(
      yb, wpb, b_proj, d_out, M, C, C);
}
